// GraphEncoder_46136538693911
// MI455X (gfx1250) — matrix operand order repaired and checked by reading, not compiled, not run on hardware
//
#include <hip/hip_runtime.h>

typedef __attribute__((ext_vector_type(16))) _Float16 v16h;
typedef __attribute__((ext_vector_type(8)))  _Float16 v8h;
typedef __attribute__((ext_vector_type(8)))  float    v8f;

#define D 128

// ---------------- utility ----------------
__global__ __launch_bounds__(256) void k_zero(float* __restrict__ p, long n) {
  long i = (long)blockIdx.x * blockDim.x + threadIdx.x;
  if (i < n) p[i] = 0.0f;
}

// deg[dst] += 1 per edge (hardware f32 atomic at L2)
__global__ __launch_bounds__(256) void k_deg(const int* __restrict__ dst,
                                             float* __restrict__ deg, int E) {
  int e = blockIdx.x * blockDim.x + threadIdx.x;
  if (e < E) unsafeAtomicAdd(&deg[dst[e]], 1.0f);
}

// in-place: deg -> rsqrt(deg + 1)   (self-loop per PyG gcn_norm)
__global__ __launch_bounds__(256) void k_rsqrt(float* __restrict__ deg, int N) {
  int n = blockIdx.x * blockDim.x + threadIdx.x;
  if (n < N) deg[n] = rsqrtf(deg[n] + 1.0f);
}

// ---------------- WMMA GEMM:  Out[M,128] = A[M,128] @ W[128,128] (+bias) ----------------
// One workgroup = 16 output rows x 128 cols; M must be a multiple of 16 (grid = M/16).
// 8 waves, each wave owns one 16x16 tile; K-loop = 4 x v_wmma_f32_16x16x32_f16, f32 acc.
template <bool HAS_BIAS>
__global__ __launch_bounds__(256) void k_gemm(const float* __restrict__ A,
                                              const float* __restrict__ W,   // [K=128][N=128] row-major
                                              const float* __restrict__ bias,
                                              float* __restrict__ Out) {
  __shared__ _Float16 sA[16][D];        // [m][k]  row stride 256B (16B aligned chunks)
  __shared__ _Float16 sWt[D][D + 8];    // W transposed: [n][k], +8 halves pad (keeps 16B align)

  const int tid  = threadIdx.x;
  const int row0 = blockIdx.x * 16;

  // stage W^T as f16 (reused by all 8 waves)
  for (int i = tid; i < D * D; i += 256) {
    int k = i >> 7, n = i & (D - 1);
    sWt[n][k] = (_Float16)W[i];
  }
  // stage the 16-row A tile as f16
  for (int i = tid; i < 16 * D; i += 256) {
    int m = i >> 7, k = i & (D - 1);
    sA[m][k] = (_Float16)A[(size_t)(row0 + m) * D + k];
  }
  __syncthreads();

  const int wave = tid >> 5;            // 0..7 -> N tile
  const int lane = tid & 31;
  const int mrow = lane & 15;           // A fragment row
  const int hiHalf = lane >> 4;         // 0 for lanes 0-15, 1 for lanes 16-31
  const int kbA = hiHalf * 8;           // A: K base 0 or 8 (second chunk at +16)
  const int kbB = hiHalf * 8;           // B: K base 0 or 8 (second chunk at +16)
  const int ncol = (wave << 4) + (lane & 15);

  v8f acc = {};
  #pragma unroll
  for (int kk = 0; kk < D; kk += 32) {
    // A 16x32 f16 fragment: a[0..7]=K(kk+kbA..+7), a[8..15]=K(kk+16+kbA..+7)
    v8h alo = *(const v8h*)&sA[mrow][kk + kbA];
    v8h ahi = *(const v8h*)&sA[mrow][kk + 16 + kbA];
    v16h a = __builtin_shufflevector(alo, ahi, 0,1,2,3,4,5,6,7,8,9,10,11,12,13,14,15);
    // B 32x16 f16 fragment: lane holds column ncol, b[0..7]=K(kk+kbB..+7), b[8..15]=K(kk+16+kbB..+7)
    v8h blo = *(const v8h*)&sWt[ncol][kk + kbB];
    v8h bhi = *(const v8h*)&sWt[ncol][kk + kbB + 16];
    v16h b = __builtin_shufflevector(blo, bhi, 0,1,2,3,4,5,6,7,8,9,10,11,12,13,14,15);
    acc = __builtin_amdgcn_wmma_f32_16x16x32_f16(
        /*neg_a=*/false, a, /*neg_b=*/false, b,
        /*c_mod=*/(short)0, acc, /*reuse_a=*/false, /*reuse_b=*/false);
  }

  // D layout: VGPR r -> M = r + 8*hiHalf, N = lane&15 (within tile)
  float bv = 0.0f;
  if constexpr (HAS_BIAS) bv = bias[ncol];
  float* outp = Out + (size_t)(row0 + (hiHalf << 3)) * D + ncol;
  #pragma unroll
  for (int r = 0; r < 8; ++r) {
    float v = acc[r];
    if constexpr (HAS_BIAS) v += bv;
    outp[(size_t)r * D] = v;
  }
}

// ---------------- edge aggregation: agg[dst] += xw[src] * dis[src]*dis[dst] ----------------
// one wave per edge; 32 lanes x float4 = 128 floats; scatter via HW f32 atomics (L2-resident)
__global__ __launch_bounds__(256) void k_edge(const float* __restrict__ xw,
                                              const int* __restrict__ src,
                                              const int* __restrict__ dst,
                                              const float* __restrict__ dis,
                                              float* __restrict__ agg, int E) {
  int e = blockIdx.x * (blockDim.x >> 5) + (threadIdx.x >> 5);
  if (e >= E) return;
  int lane = threadIdx.x & 31;
  int s = src[e], d = dst[e];
  float norm = dis[s] * dis[d];
  const float4 v = ((const float4*)(xw + (size_t)s * D))[lane];
  float* pd = agg + (size_t)d * D + lane * 4;
  unsafeAtomicAdd(pd + 0, v.x * norm);
  unsafeAtomicAdd(pd + 1, v.y * norm);
  unsafeAtomicAdd(pd + 2, v.z * norm);
  unsafeAtomicAdd(pd + 3, v.w * norm);
}

// h1 = relu(agg + xw*dis^2 + b)
__global__ __launch_bounds__(256) void k_self_relu(const float* __restrict__ agg,
                                                   const float* __restrict__ xw,
                                                   const float* __restrict__ dis,
                                                   const float* __restrict__ b,
                                                   float* __restrict__ h, long ND) {
  long i = (long)blockIdx.x * blockDim.x + threadIdx.x;
  if (i < ND) {
    int n = (int)(i >> 7), c = (int)(i & (D - 1));
    float d2 = dis[n] * dis[n];
    float v = agg[i] + xw[i] * d2 + b[c];
    h[i] = fmaxf(v, 0.0f);
  }
}

// h2 = (agg + xw*dis^2 + b) + h1 ; gsum[batch[n]][c] += h2   (h2 never materialized)
__global__ __launch_bounds__(256) void k_pool(const float* __restrict__ agg,
                                              const float* __restrict__ xw,
                                              const float* __restrict__ dis,
                                              const float* __restrict__ b,
                                              const float* __restrict__ h1,
                                              const int* __restrict__ batch,
                                              float* __restrict__ gsum, long ND) {
  long i = (long)blockIdx.x * blockDim.x + threadIdx.x;
  if (i < ND) {
    int n = (int)(i >> 7), c = (int)(i & (D - 1));
    float d2 = dis[n] * dis[n];
    float v = agg[i] + xw[i] * d2 + b[c] + h1[i];
    unsafeAtomicAdd(&gsum[(size_t)batch[n] * D + c], v);
  }
}

__global__ __launch_bounds__(256) void k_cnt(const int* __restrict__ batch,
                                             float* __restrict__ cnt, int N) {
  int n = blockIdx.x * blockDim.x + threadIdx.x;
  if (n < N) unsafeAtomicAdd(&cnt[batch[n]], 1.0f);
}

__global__ __launch_bounds__(256) void k_mean(const float* __restrict__ gsum,
                                              const float* __restrict__ cnt,
                                              float* __restrict__ gmean, int GD) {
  int i = blockIdx.x * blockDim.x + threadIdx.x;
  if (i < GD) {
    int g = i >> 7;
    gmean[i] = gsum[i] / fmaxf(cnt[g], 1.0f);
  }
}

// ---------------- launch ----------------
static inline int cdiv(long a, long b) { return (int)((a + b - 1) / b); }

extern "C" void kernel_launch(void* const* d_in, const int* in_sizes, int n_in,
                              void* d_out, int out_size, void* d_ws, size_t ws_size,
                              hipStream_t stream) {
  const float* x     = (const float*)d_in[0];
  const int*   ei    = (const int*)d_in[1];
  const int*   batch = (const int*)d_in[2];
  const float* W1    = (const float*)d_in[3];
  const float* b1    = (const float*)d_in[4];
  const float* W2    = (const float*)d_in[5];
  const float* b2    = (const float*)d_in[6];
  const float* Wout  = (const float*)d_in[7];
  const float* bout  = (const float*)d_in[8];
  float* out = (float*)d_out;

  const int N = in_sizes[0] / D;   // 100000  (multiple of 16)
  const int E = in_sizes[1] / 2;   // 1600000
  const int G = out_size / D;      // 64      (multiple of 16)
  const int* src = ei;
  const int* dst = ei + E;
  const long ND = (long)N * D;

  float* dis   = (float*)d_ws;                 // N   (deg -> rsqrt in place)
  float* xw    = dis  + N;                     // N*D (xw1 then xw2)
  float* agg   = xw   + ND;                    // N*D (reused)
  float* h1    = agg  + ND;                    // N*D
  float* gsum  = h1   + ND;                    // G*D
  float* cnt   = gsum + (size_t)G * D;         // G
  float* gmean = cnt  + G;                     // G*D

  dim3 b256(256);

  // ---- degree norm ----
  k_zero <<<cdiv(N, 256), b256, 0, stream>>>(dis, N);
  k_deg  <<<cdiv(E, 256), b256, 0, stream>>>(dst, dis, E);
  k_rsqrt<<<cdiv(N, 256), b256, 0, stream>>>(dis, N);

  // ---- layer 1: h1 = relu(gcn_conv(x, W1, b1)) ----
  k_gemm<false><<<N / 16, b256, 0, stream>>>(x, W1, nullptr, xw);
  k_zero <<<cdiv(ND, 256), b256, 0, stream>>>(agg, ND);
  k_edge <<<cdiv(E, 8), b256, 0, stream>>>(xw, src, dst, dis, agg, E);
  k_self_relu<<<cdiv(ND, 256), b256, 0, stream>>>(agg, xw, dis, b1, h1, ND);

  // ---- layer 2 + residual + mean-pool (fused epilogue) ----
  k_gemm<false><<<N / 16, b256, 0, stream>>>(h1, W2, nullptr, xw);
  k_zero <<<cdiv(ND, 256), b256, 0, stream>>>(agg, ND);
  k_edge <<<cdiv(E, 8), b256, 0, stream>>>(xw, src, dst, dis, agg, E);
  k_zero <<<cdiv((long)G * D + G, 256), b256, 0, stream>>>(gsum, (long)G * D + G);
  k_pool <<<cdiv(ND, 256), b256, 0, stream>>>(agg, xw, dis, b2, h1, batch, gsum, ND);
  k_cnt  <<<cdiv(N, 256), b256, 0, stream>>>(batch, cnt, N);
  k_mean <<<cdiv((long)G * D, 256), b256, 0, stream>>>(gsum, cnt, gmean, G * D);

  // ---- readout: out = gmean @ Wout + bout ----
  k_gemm<true><<<G / 16, b256, 0, stream>>>(gmean, Wout, bout, out);
}
